// GNN_45354854646409
// MI455X (gfx1250) — matrix operand order repaired and checked by reading, not compiled, not run on hardware
//
#include <hip/hip_runtime.h>
#include <hip/hip_bf16.h>

// ---------------- problem constants ----------------
#define NN      50000
#define EE      400000
#define GG      64
#define EPSV    1e-5f

typedef _Float16 v16h __attribute__((ext_vector_type(16)));
typedef _Float16 v8h  __attribute__((ext_vector_type(8)));
typedef float    v8f  __attribute__((ext_vector_type(8)));

// Padded WMMA-B fragment buffer: per chunk, lane stride = 24 halfs (48 B) so the
// two ds_load_b128 per lane hit every bank exactly twice (minimal 2-pass).
#define W2PP_LANE_H   24u                    // halfs per lane slot (16 data + 8 pad)
#define W2PP_CHUNK_H  (32u * W2PP_LANE_H)    // 768 halfs per chunk
#define W2PP_USED_H   (17u * W2PP_CHUNK_H)   // 13056 halfs
#define W2PP_STRIDE_H 14336u                 // padded to 7*256*16B copy units (28,672 B)

// ---------------- workspace layout (float offsets) ----------------
// accumulator region (zeroed every launch):
#define OFF_MOM    0u                       // 44 used (8 mean + 36 upper-tri 2nd moments)
#define OFF_SSUM   72u                      // 64: [sum1 16][sq1 16][sum2 16][sq2 16]
#define OFF_POOL   136u                     // 1024: pooled [G,16]
#define OFF_GCNT   1160u                    // 64: nodes per graph
#define OFF_CNT    1224u                    // NN: in-degree counts
#define OFF_AGG1   (1224u + (unsigned)NN)             // NN*16 (agg1 -> h1 -> h_act, in place)
#define OFF_AGG2   (OFF_AGG1 + (unsigned)NN * 16u)    // NN*16 (agg2 -> h2, in place)
#define ZERO_TOT   (OFF_AGG2 + (unsigned)NN * 16u)
// non-accumulator region (fully overwritten every launch):
#define OFF_EAFF   ZERO_TOT                 // 128: edge BN affine [mlp][aA 32 | aC 32]
#define OFF_NAFF   (ZERO_TOT + 128u)        // 64:  node BN affine [a1 16|c1 16|a2 16|c2 16]
#define OFF_W2PP   (ZERO_TOT + 192u)        // 2 * W2PP_STRIDE_H halfs (16B-aligned)

// =====================================================================
// zero the accumulator region
// =====================================================================
__global__ __launch_bounds__(256) void k_zero(float* __restrict__ w, unsigned total) {
    unsigned i = blockIdx.x * 256u + threadIdx.x;
    if (i < total) w[i] = 0.0f;
}

// =====================================================================
// Pre-permute W2 (+ b2 as virtual 17th K-chunk) into padded WMMA-B order.
// B fragment (f16, 32x16): lane l holds column n=l&15; element t <-> localK=(t<8 ? 8*(l>>4)+t : 16+8*(l>>4)+(t-8)).
// z index K = k*16+i (k: hidden 0..31, i: src channel 0..15).
// =====================================================================
__global__ __launch_bounds__(256)
void k_prep_w2(const float* __restrict__ W2a, const float* __restrict__ b2a,
               const float* __restrict__ W2b, const float* __restrict__ b2b,
               _Float16* __restrict__ w2pp) {
    unsigned id = blockIdx.x * 256u + threadIdx.x;
    if (id >= 2u * W2PP_STRIDE_H) return;
    unsigned mlp = id / W2PP_STRIDE_H;
    unsigned r   = id % W2PP_STRIDE_H;
    float v = 0.0f;
    if (r < W2PP_USED_H) {
        unsigned c  = r / W2PP_CHUNK_H;
        unsigned rr = r % W2PP_CHUNK_H;
        unsigned l  = rr / W2PP_LANE_H;
        unsigned t  = rr % W2PP_LANE_H;
        if (t < 16u) {
            unsigned hi = l >> 4, n = l & 15u;
            unsigned localK = (t < 8u) ? hi * 8u + t : 16u + hi * 8u + (t - 8u);
            const float* W2 = mlp ? W2b : W2a;
            const float* b2 = mlp ? b2b : b2a;
            if (c < 16u) {
                unsigned K = c * 32u + localK;
                v = W2[(K >> 4) * 256u + (K & 15u) * 16u + n];
            } else {
                v = (localK < 16u) ? b2[localK * 16u + n] : 0.0f;  // bias row (hid==1)
            }
        }
    }
    w2pp[id] = (_Float16)v;
}

// =====================================================================
// One pass over edge_attr: mean (8) + upper-tri second moments (36).
// Shuffle-reduce per wave, LDS-combine per block, 44 global atomics/block.
// =====================================================================
__global__ __launch_bounds__(256) void k_edge_stats(const float* __restrict__ ea, float* __restrict__ mom) {
    __shared__ float sAcc[48];
    int tid = threadIdx.x;
    if (tid < 44) sAcc[tid] = 0.0f;
    __syncthreads();

    float m1[8];  float m2[36];
#pragma unroll
    for (int j = 0; j < 8; ++j)  m1[j] = 0.0f;
#pragma unroll
    for (int k = 0; k < 36; ++k) m2[k] = 0.0f;

    for (unsigned e = blockIdx.x * 256u + (unsigned)tid; e < (unsigned)EE; e += gridDim.x * 256u) {
        const float4* p = reinterpret_cast<const float4*>(ea + (size_t)e * 8u);
        float4 a = p[0], b = p[1];
        float v[8] = {a.x, a.y, a.z, a.w, b.x, b.y, b.z, b.w};
#pragma unroll
        for (int j = 0; j < 8; ++j) m1[j] += v[j];
        int idx = 0;
#pragma unroll
        for (int j = 0; j < 8; ++j)
#pragma unroll
            for (int jp = j; jp < 8; ++jp) { m2[idx] += v[j] * v[jp]; ++idx; }
    }
#pragma unroll
    for (int off = 16; off; off >>= 1) {
#pragma unroll
        for (int j = 0; j < 8; ++j)  m1[j] += __shfl_xor(m1[j], off, 32);
#pragma unroll
        for (int k = 0; k < 36; ++k) m2[k] += __shfl_xor(m2[k], off, 32);
    }
    if ((tid & 31) == 0) {
#pragma unroll
        for (int j = 0; j < 8; ++j)  atomicAdd(&sAcc[j], m1[j]);
#pragma unroll
        for (int k = 0; k < 36; ++k) atomicAdd(&sAcc[8 + k], m2[k]);
    }
    __syncthreads();
    if (tid < 44) atomicAdd(&mom[tid], sAcc[tid]);
}

// =====================================================================
// BN(edge-hidden) affine from moments:  var_c = w^T M w - (w^T mu)^2
// h_bn = aA * (ea@W1) + aC   (absorbs b1, gamma, beta, mean, var)
// =====================================================================
__global__ void k_edge_stats_fin(const float* __restrict__ mom,
                                 const float* __restrict__ W1a, const float* __restrict__ ga, const float* __restrict__ bea,
                                 const float* __restrict__ W1b, const float* __restrict__ gb, const float* __restrict__ beb,
                                 float* __restrict__ aff) {
    int t = threadIdx.x;
    if (t >= 64) return;
    int mlp = t >> 5, col = t & 31;
    const float* W1 = mlp ? W1b : W1a;
    const float* g  = mlp ? gb  : ga;
    const float* be = mlp ? beb : bea;
    const float invE = 1.0f / (float)EE;
    float w[8];
#pragma unroll
    for (int j = 0; j < 8; ++j) w[j] = W1[j * 32 + col];
    float umean = 0.0f;
#pragma unroll
    for (int j = 0; j < 8; ++j) umean += mom[j] * invE * w[j];
    float q = 0.0f; int idx = 0;
#pragma unroll
    for (int j = 0; j < 8; ++j)
#pragma unroll
        for (int jp = j; jp < 8; ++jp) {
            float term = w[j] * (mom[8 + idx] * invE) * w[jp];
            q += (jp == j) ? term : 2.0f * term;
            ++idx;
        }
    float var = fmaxf(q - umean * umean, 0.0f);
    float inv = 1.0f / sqrtf(var + EPSV);
    float aA = g[col] * inv;
    aff[mlp * 64 + col]      = aA;
    aff[mlp * 64 + 32 + col] = be[col] - aA * umean;
}

// =====================================================================
// Fused NNConv edge kernel: edge-MLP (inline BN+ReLU) -> outer-product z
// -> 17x V_WMMA_F32_16X16X32_F16 (B from LDS, staged by async-to-LDS)
// -> scatter-atomic into agg.  One wave per 16-edge tile, 8 tiles/block.
// =====================================================================
template <bool WRITE_CNT>
__global__ __launch_bounds__(256)
void k_conv_edges(const float* __restrict__ xsrc,       // [N,16]
                  const float* __restrict__ ea,         // [E,8]
                  const int*   __restrict__ eidx,       // [2,E]
                  const float* __restrict__ W1,         // [8,32]
                  const float* __restrict__ aff,        // [aA 32 | aC 32]
                  const _Float16* __restrict__ w2pp,    // [W2PP_STRIDE_H], padded B fragments
                  float* __restrict__ agg,              // [N,16]
                  float* __restrict__ cnt)              // [N]
{
    __attribute__((aligned(64))) __shared__ _Float16 sW2[W2PP_STRIDE_H];  // 28 KB
    __shared__ float sW1[256];
    __shared__ float sAf[64];
    int tid = threadIdx.x;
    sW1[tid] = W1[tid];
    if (tid < 64) sAf[tid] = aff[tid];

    // Async-copy the pre-swizzled W2 fragments into LDS (7 x 16 B per thread,
    // exact). ASYNCcnt is per-wave; each wave drains its own, barrier publishes.
    {
        unsigned           ldsBase = (unsigned)(size_t)(&sW2[0]);
        unsigned long long gBase   = (unsigned long long)(size_t)w2pp;
#pragma unroll
        for (int it = 0; it < 7; ++it) {
            unsigned           off = ((unsigned)it * 256u + (unsigned)tid) * 16u;
            unsigned           la  = ldsBase + off;
            unsigned long long ga  = gBase + off;
            asm volatile("global_load_async_to_lds_b128 %0, %1, off" :: "v"(la), "v"(ga) : "memory");
        }
#if __has_builtin(__builtin_amdgcn_s_wait_asynccnt)
        __builtin_amdgcn_s_wait_asynccnt(0);
#else
        asm volatile("s_wait_asynccnt 0x0" ::: "memory");
#endif
    }
    __syncthreads();

    int lane = tid & 31;
    int wv   = tid >> 5;
    int e0   = (blockIdx.x * 8 + wv) * 16;
    if (e0 >= EE) return;                       // uniform per wave; grid is exact anyway

    int m   = lane & 15;                        // edge-in-tile (A-matrix row M)
    int hi  = lane >> 4;
    int e   = e0 + m;
    int src = eidx[e];
    int dst = eidx[EE + e];

    const float4* eap = reinterpret_cast<const float4*>(ea + (size_t)e * 8u);
    float4 ea0 = eap[0], ea1 = eap[1];
    float eav[8] = {ea0.x, ea0.y, ea0.z, ea0.w, ea1.x, ea1.y, ea1.z, ea1.w};

    // this lane's half of the 32 hidden units (k = 16*hi + q)
    float acc[16];
#pragma unroll
    for (int q = 0; q < 16; ++q) acc[q] = 0.0f;
#pragma unroll
    for (int j = 0; j < 8; ++j) {
        const float* wr = &sW1[j * 32 + hi * 16];
        float v = eav[j];
#pragma unroll
        for (int q = 0; q < 16; ++q) acc[q] += v * wr[q];
    }
    float own[16];
#pragma unroll
    for (int q = 0; q < 16; ++q)
        own[q] = fmaxf(sAf[hi * 16 + q] * acc[q] + sAf[32 + hi * 16 + q], 0.0f);

    // exchange halves across lane^16 so every lane holds all 32 hid of its edge
    _Float16 hidAll[32];
#pragma unroll
    for (int q = 0; q < 16; ++q) {
        float other = __shfl_xor(own[q], 16, 32);
        hidAll[q]      = (_Float16)(hi ? other  : own[q]);
        hidAll[16 + q] = (_Float16)(hi ? own[q] : other);
    }

    // gather source features: channels hi*8 .. hi*8+7 (matches A-fragment i-range)
    const float4* xp = reinterpret_cast<const float4*>(xsrc + (size_t)src * 16u + (unsigned)hi * 8u);
    float4 x0 = xp[0], x1 = xp[1];
    _Float16 xs8[8] = {(_Float16)x0.x, (_Float16)x0.y, (_Float16)x0.z, (_Float16)x0.w,
                       (_Float16)x1.x, (_Float16)x1.y, (_Float16)x1.z, (_Float16)x1.w};

    // msg[16x16] = z[16x(512+16)] @ W2pp : 17 chained WMMAs (chunk 16 = b2 bias row)
    v8f d = {0.f, 0.f, 0.f, 0.f, 0.f, 0.f, 0.f, 0.f};
#pragma unroll
    for (int c = 0; c < 17; ++c) {
        _Float16 hA = (c < 16) ? hidAll[2 * c]     : (_Float16)1.0f;
        _Float16 hB = (c < 16) ? hidAll[2 * c + 1] : (_Float16)0.0f;
        v16h a;
#pragma unroll
        for (int t = 0; t < 8; ++t) { a[t] = hA * xs8[t]; a[8 + t] = hB * xs8[t]; }
        const _Float16* bp = &sW2[(unsigned)c * W2PP_CHUNK_H + (unsigned)lane * W2PP_LANE_H];
        v8h b0 = *reinterpret_cast<const v8h*>(bp);
        v8h b1 = *reinterpret_cast<const v8h*>(bp + 8);
        v16h b;
#pragma unroll
        for (int t = 0; t < 8; ++t) { b[t] = b0[t]; b[8 + t] = b1[t]; }
        d = __builtin_amdgcn_wmma_f32_16x16x32_f16(false, a, false, b, (short)0, d, false, false);
    }

    // D layout: lane l, vgpr r -> edge 8*hi + r, channel l&15. Scatter-add.
#pragma unroll
    for (int r = 0; r < 8; ++r) {
        int dr = __shfl(dst, hi * 8 + r, 32);
        atomicAdd(&agg[(size_t)dr * 16u + (unsigned)m], d[r]);
    }
    if (WRITE_CNT && lane < 16) atomicAdd(&cnt[dst], 1.0f);
}

// =====================================================================
// Root transform + mean-normalize agg (in place) + block-reduced BN stats.
// 256 threads = 16 nodes x 16 channels; node features staged in LDS.
// =====================================================================
__global__ __launch_bounds__(256)
void k_conv_root(float* __restrict__ agg_h, const float* __restrict__ cnt,
                 const float* __restrict__ xin, const float* __restrict__ root,
                 const float* __restrict__ bias,
                 float* __restrict__ statSum, float* __restrict__ statSq) {
    __shared__ float sRoot[256];
    __shared__ float sBias[16];
    __shared__ float sX[256];
    __shared__ float s1[256], s2[256];
    int tid = threadIdx.x;
    sRoot[tid] = root[tid];
    if (tid < 16) sBias[tid] = bias[tid];
    unsigned gi = blockIdx.x * 256u + (unsigned)tid;       // block covers 16 nodes x 16 feats
    sX[tid] = (gi < (unsigned)NN * 16u) ? xin[gi] : 0.0f;
    __syncthreads();

    int ch = tid & 15, ni = tid >> 4;
    unsigned n = blockIdx.x * 16u + (unsigned)ni;
    float v = 0.0f;
    if (n < (unsigned)NN) {
        float c = fmaxf(cnt[n], 1.0f);
        float r = sBias[ch];
#pragma unroll
        for (int i = 0; i < 16; ++i) r += sX[ni * 16 + i] * sRoot[i * 16 + ch];
        v = agg_h[(size_t)n * 16u + ch] / c + r;
        agg_h[(size_t)n * 16u + ch] = v;
    }
    s1[tid] = v; s2[tid] = v * v;
    __syncthreads();
#pragma unroll
    for (int off = 128; off >= 16; off >>= 1) {
        if (tid < off) { s1[tid] += s1[tid + off]; s2[tid] += s2[tid + off]; }
        __syncthreads();
    }
    if (tid < 16) { atomicAdd(&statSum[tid], s1[tid]); atomicAdd(&statSq[tid], s2[tid]); }
}

__global__ void k_node_bn_fin(const float* __restrict__ ssum, const float* __restrict__ ssq,
                              const float* __restrict__ g, const float* __restrict__ b,
                              float* __restrict__ naff, float invN) {
    int t = threadIdx.x;
    if (t >= 16) return;
    float mean = ssum[t] * invN;
    float var  = fmaxf(ssq[t] * invN - mean * mean, 0.0f);
    float inv  = 1.0f / sqrtf(var + EPSV);
    float a = g[t] * inv;
    naff[t] = a;
    naff[16 + t] = b[t] - a * mean;
}

__global__ __launch_bounds__(256) void k_bn_relu(float* __restrict__ h, const float* __restrict__ naff) {
    unsigned i = blockIdx.x * 256u + threadIdx.x;
    if (i >= (unsigned)NN * 16u) return;
    unsigned ch = i & 15u;
    h[i] = fmaxf(naff[ch] * h[i] + naff[16 + ch], 0.0f);
}

__global__ __launch_bounds__(256)
void k_pool(const float* __restrict__ h2, const float* __restrict__ naff,
            const int* __restrict__ batch, float* __restrict__ pooled, float* __restrict__ gcnt) {
    unsigned i = blockIdx.x * 256u + threadIdx.x;
    if (i >= (unsigned)NN * 16u) return;
    unsigned n = i >> 4, ch = i & 15u;
    float v = fmaxf(naff[ch] * h2[i] + naff[16 + ch], 0.0f);
    int b = batch[n];
    atomicAdd(&pooled[b * 16 + (int)ch], v);
    if (ch == 0) atomicAdd(&gcnt[b], 1.0f);
}

__global__ void k_head(const float* __restrict__ pooled, const float* __restrict__ gcnt,
                       const float* __restrict__ fc1W, const float* __restrict__ fc1b,
                       const float* __restrict__ fc2W, const float* __restrict__ fc2b,
                       float* __restrict__ out) {
    int g = threadIdx.x;
    if (g >= GG) return;
    float inv = 1.0f / fmaxf(gcnt[g], 1.0f);
    float p[16];
#pragma unroll
    for (int i = 0; i < 16; ++i) p[i] = pooled[g * 16 + i] * inv;
    float hd[16];
#pragma unroll
    for (int j = 0; j < 16; ++j) {
        float s = fc1b[j];
#pragma unroll
        for (int i = 0; i < 16; ++i) s += p[i] * fc1W[i * 16 + j];
        hd[j] = fmaxf(s, 0.0f);
    }
#pragma unroll
    for (int t = 0; t < 2; ++t) {
        float o = fc2b[t];
#pragma unroll
        for (int j = 0; j < 16; ++j) o += hd[j] * fc2W[j * 2 + t];
        out[g * 2 + t] = o;
    }
}

// =====================================================================
extern "C" void kernel_launch(void* const* d_in, const int* in_sizes, int n_in,
                              void* d_out, int out_size, void* d_ws, size_t ws_size,
                              hipStream_t stream) {
    (void)in_sizes; (void)n_in; (void)out_size; (void)ws_size;
    const float* x        = (const float*)d_in[0];
    const float* ea       = (const float*)d_in[1];
    const float* e1_W1    = (const float*)d_in[2];
    const float* e1_g     = (const float*)d_in[4];
    const float* e1_be    = (const float*)d_in[5];
    const float* e1_W2    = (const float*)d_in[6];
    const float* e1_b2    = (const float*)d_in[7];
    const float* c1_root  = (const float*)d_in[8];
    const float* c1_bias  = (const float*)d_in[9];
    const float* e2_W1    = (const float*)d_in[10];
    const float* e2_g     = (const float*)d_in[12];
    const float* e2_be    = (const float*)d_in[13];
    const float* e2_W2    = (const float*)d_in[14];
    const float* e2_b2    = (const float*)d_in[15];
    const float* c2_root  = (const float*)d_in[16];
    const float* c2_bias  = (const float*)d_in[17];
    const float* bn1_g    = (const float*)d_in[18];
    const float* bn1_b    = (const float*)d_in[19];
    const float* bn2_g    = (const float*)d_in[20];
    const float* bn2_b    = (const float*)d_in[21];
    const float* fc1_W    = (const float*)d_in[22];
    const float* fc1_b    = (const float*)d_in[23];
    const float* fc2_W    = (const float*)d_in[24];
    const float* fc2_b    = (const float*)d_in[25];
    const int*   eidx     = (const int*)d_in[26];
    const int*   batch    = (const int*)d_in[27];

    float* w      = (float*)d_ws;
    float* mom    = w + OFF_MOM;
    float* ssum1  = w + OFF_SSUM;
    float* ssq1   = ssum1 + 16;
    float* ssum2  = ssum1 + 32;
    float* ssq2   = ssum1 + 48;
    float* pooled = w + OFF_POOL;
    float* gcnt   = w + OFF_GCNT;
    float* cnt    = w + OFF_CNT;
    float* agg1   = w + OFF_AGG1;   // -> h1 -> h_act (in place)
    float* agg2   = w + OFF_AGG2;   // -> h2 (in place)
    float* eaff   = w + OFF_EAFF;
    float* naff   = w + OFF_NAFF;
    _Float16* w2pp = (_Float16*)(w + OFF_W2PP);

    k_zero<<<(ZERO_TOT + 255u) / 256u, 256, 0, stream>>>(w, ZERO_TOT);
    k_prep_w2<<<(2u * W2PP_STRIDE_H + 255u) / 256u, 256, 0, stream>>>(e1_W2, e1_b2, e2_W2, e2_b2, w2pp);
    k_edge_stats<<<512, 256, 0, stream>>>(ea, mom);
    k_edge_stats_fin<<<1, 64, 0, stream>>>(mom, e1_W1, e1_g, e1_be, e2_W1, e2_g, e2_be, eaff);

    // conv1: 25000 tiles of 16 edges, 8 waves (tiles) per block
    k_conv_edges<true><<<EE / 128, 256, 0, stream>>>(x, ea, eidx, e1_W1, eaff, w2pp, agg1, cnt);
    k_conv_root<<<(NN + 15) / 16, 256, 0, stream>>>(agg1, cnt, x, c1_root, c1_bias, ssum1, ssq1);
    k_node_bn_fin<<<1, 32, 0, stream>>>(ssum1, ssq1, bn1_g, bn1_b, naff, 1.0f / (float)NN);
    k_bn_relu<<<((unsigned)NN * 16u + 255u) / 256u, 256, 0, stream>>>(agg1, naff);

    // conv2 (reuses in-degree counts)
    k_conv_edges<false><<<EE / 128, 256, 0, stream>>>(agg1, ea, eidx, e2_W1, eaff + 64,
                                                      w2pp + W2PP_STRIDE_H, agg2, cnt);
    k_conv_root<<<(NN + 15) / 16, 256, 0, stream>>>(agg2, cnt, agg1, c2_root, c2_bias, ssum2, ssq2);
    k_node_bn_fin<<<1, 32, 0, stream>>>(ssum2, ssq2, bn2_g, bn2_b, naff + 32, 1.0f / (float)NN);

    k_pool<<<((unsigned)NN * 16u + 255u) / 256u, 256, 0, stream>>>(agg2, naff + 32, batch, pooled, gcnt);
    k_head<<<1, 64, 0, stream>>>(pooled, gcnt, fc1_W, fc1_b, fc2_W, fc2_b, (float*)d_out);
}
